// Tacotron2_23167053595283
// MI455X (gfx1250) — compile-verified, matrix operand order checked by reading, not run on hardware
//
#include <hip/hip_runtime.h>

typedef __bf16 bf16;
typedef __attribute__((ext_vector_type(16))) __bf16 v16bf;
typedef __attribute__((ext_vector_type(8)))  __bf16 v8bf;
typedef __attribute__((ext_vector_type(8)))  float   v8f;

#define ZV8F {0.f,0.f,0.f,0.f,0.f,0.f,0.f,0.f}
#define WMMA_BF(a,b,c) __builtin_amdgcn_wmma_f32_16x16x32_bf16(false,(a),false,(b),(short)0,(c),false,false)
#define GSL(i,n) for (long i = (long)blockIdx.x*blockDim.x + threadIdx.x; i < (n); i += (long)gridDim.x*blockDim.x)

__device__ __forceinline__ float sigm(float x){ return 1.f/(1.f+__expf(-x)); }

// Load a 16x32 bf16 tile (rows x K) into WMMA A/B operand layout.
// lane<16: row=lane, K 0..7 (v0-3) and 16..23 (v4-7); lane>=16: row=lane-16, K 8..15 / 24..31.
__device__ __forceinline__ v16bf ld_tile(const bf16* p, int ld){
  int lane = threadIdx.x & 31;
  const bf16* q = p + (size_t)(lane & 15) * ld + ((lane & 16) ? 8 : 0);
  v8bf lo = *(const v8bf*)(q);
  v8bf hi = *(const v8bf*)(q + 16);
  v16bf v;
#pragma unroll
  for (int i = 0; i < 8; ++i){ v[i] = lo[i]; v[i+8] = hi[i]; }
  return v;
}

// ---------------- elementwise / packing kernels ----------------
__global__ void zero_k(unsigned* p, long n){ GSL(i, n) p[i] = 0u; }

__global__ void cvt_k(bf16* d, const float* s, long n){ GSL(i, n) d[i] = (bf16)s[i]; }

__global__ void cvtpad_k(bf16* d, const float* s, int N, int K, int Kp){
  GSL(i, (long)N*Kp){ int r = (int)(i / Kp), c = (int)(i % Kp);
    d[i] = (c < K) ? (bf16)s[(size_t)r*K + c] : (bf16)0.f; }
}

// src [O][I][KT] fp32 -> dst [KT][Op][Ip] bf16 (zero padded)
__global__ void packconv_k(bf16* d, const float* s, int O, int I, int KT, int Op, int Ip){
  GSL(i, (long)KT*Op*Ip){
    int k = (int)(i / ((long)Op*Ip)); long rem = i - (long)k*Op*Ip;
    int o = (int)(rem / Ip), c = (int)(rem % Ip);
    float v = (o < O && c < I) ? s[((size_t)o*I + c)*KT + k] : 0.f;
    d[i] = (bf16)v; }
}

// dst [N][Ka+Kb] bf16 = concat of a [N][Ka], b [N][Kb]
__global__ void concat2_k(bf16* d, const float* a, int Ka, const float* b, int Kb, int N){
  int Kt = Ka + Kb;
  GSL(i, (long)N*Kt){ int r = (int)(i / Kt), c = (int)(i % Kt);
    d[i] = (bf16)((c < Ka) ? a[(size_t)r*Ka + c] : b[(size_t)r*Kb + (c - Ka)]); }
}

__global__ void addb_k(float* d, const float* a, const float* b, long n){ GSL(i, n) d[i] = a[i] + b[i]; }

__global__ void convsc_k(float* cb, float* sc, float* bt, const float* b, const float* g,
                         const float* be, int O, int Op){
  GSL(i, (long)Op){
    cb[i] = (i < O) ? b[i] : 0.f;
    sc[i] = (i < O) ? g[i] * rsqrtf(1.00001f) : 0.f;
    bt[i] = (i < O) ? be[i] : 0.f; }
}

__global__ void packms_k(bf16* W, float* bias, const float* mw, const float* mb2,
                         const float* sw, const float* sb2){
  GSL(i, 96L*1536){
    int r = (int)(i / 1536), c = (int)(i % 1536);
    float v = (r < 80) ? mw[(size_t)r*1536 + c] : ((r == 80) ? sw[c] : 0.f);
    W[i] = (bf16)v;
    if (c == 0) bias[r] = (r < 80) ? mb2[r] : ((r == 80) ? sb2[0] : 0.f); }
}

__global__ void embed_k(bf16* xp, const int* x, const float* emb){
  GSL(i, 1048576L){ // 16*128*512
    int c = (int)(i & 511), t = (int)((i >> 9) & 127), b = (int)(i >> 16);
    xp[((size_t)b*132 + 2 + t)*512 + c] = (bf16)emb[(size_t)x[b*128 + t]*512 + c]; }
}

__global__ void stage_xs_k(bf16* xs, const bf16* xp){
  GSL(i, 1048576L){
    int c = (int)(i & 511), t = (int)((i >> 9) & 127), b = (int)(i >> 16);
    xs[((size_t)t*16 + b)*512 + c] = xp[((size_t)b*132 + 2 + t)*512 + c]; }
}

__global__ void stage_tf_k(bf16* tf, const float* mel){
  GSL(i, 614400L){ // 6400*96
    int c = (int)(i % 96); long r = i / 96;
    int t = (int)(r >> 4), b = (int)(r & 15);
    float v = (t == 0 || c >= 80) ? 0.f : mel[((size_t)b*400 + (t - 1))*80 + c];
    tf[i] = (bf16)v; }
}

// ---------------- generic WMMA GEMM: out = act(A @ W^T + bias) ----------------
// A: row-major [M][lda] bf16 (rows mapped through aBlk/aSkip/aOff), W: [N][K] bf16.
__global__ void gemm_k(const bf16* __restrict__ A, int lda, const bf16* __restrict__ W,
                       const float* __restrict__ bias, int K, int act,
                       float* outF, int ldf, bf16* outB, int ldb2,
                       int aBlk, int aSkip, int aOff, int oBlk, int oSkip, int oOff){
  int nt = blockIdx.x, mt = blockIdx.y;
  int m0 = mt << 4;
  int arow0 = (m0 / aBlk)*(aBlk + aSkip) + aOff + (m0 % aBlk);
  const bf16* Ab = A + (size_t)arow0 * lda;
  const bf16* Wb = W + (size_t)nt*16*K;
  v8f acc = ZV8F;
  for (int k = 0; k < K; k += 32)
    acc = WMMA_BF(ld_tile(Ab + k, lda), ld_tile(Wb + k, K), acc);
  int lane = threadIdx.x;
  int n = lane & 15, mb = (lane & 16) ? 8 : 0;
  int col = nt*16 + n;
  float bs = bias ? bias[col] : 0.f;
#pragma unroll
  for (int r = 0; r < 8; ++r){
    float v = acc[r] + bs;
    if (act == 1) v = fmaxf(v, 0.f); else if (act == 2) v = tanhf(v);
    int m = m0 + mb + r;
    int orow = (m / oBlk)*(oBlk + oSkip) + oOff + (m % oBlk);
    if (outF) outF[(size_t)orow*ldf + col] = v;
    if (outB) outB[(size_t)orow*ldb2 + col] = (bf16)v;
  }
}

// ---------------- conv1d K=5 + BN (+act) as implicit WMMA GEMM ----------------
// X: [B][T+4][Cin] bf16 (rows 0,1 and T+2,T+3 zero); W: [5][Cout][Cin] bf16.
__global__ void conv5_k(const bf16* __restrict__ X, const bf16* __restrict__ W,
                        const float* cb, const float* sc, const float* bt,
                        int Cin, int Cout, int T, int act,
                        bf16* outB, const float* addF, float* outF, int Oreal){
  int nt = blockIdx.x, mt = blockIdx.y;
  int TB = T >> 4;
  int b = mt / TB, t0 = (mt % TB) << 4;
  int Tp = T + 4;
  v8f acc = ZV8F;
  for (int k = 0; k < 5; ++k){
    const bf16* Ab = X + ((size_t)b*Tp + t0 + k)*Cin;
    const bf16* Wb = W + ((size_t)k*Cout + nt*16)*Cin;
    for (int kk = 0; kk < Cin; kk += 32)
      acc = WMMA_BF(ld_tile(Ab + kk, Cin), ld_tile(Wb + kk, Cin), acc);
  }
  int lane = threadIdx.x;
  int n = lane & 15, mb = (lane & 16) ? 8 : 0;
  int o = nt*16 + n;
  float c0 = cb[o], s0 = sc[o], b0 = bt[o];
#pragma unroll
  for (int r = 0; r < 8; ++r){
    int t = t0 + mb + r;
    float v = (acc[r] + c0)*s0 + b0;
    if (act == 1) v = fmaxf(v, 0.f); else if (act == 2) v = tanhf(v);
    size_t row = (size_t)b*Tp + 2 + t;
    if (outF){
      if (o < Oreal){
        float f = v + addF[((size_t)b*T + t)*Oreal + o];
        outF[((size_t)b*T + t)*Oreal + o] = f;
        outB[row*Cout + o] = (bf16)f;
      } else outB[row*Cout + o] = (bf16)0.f;
    } else outB[row*Cout + o] = (bf16)v;
  }
}

// ---------------- encoder BiLSTM recurrent kernel (one WG per direction) ----------------
// gx: [2][2048][1024] precomputed x@Wih^T (t-major rows t*16+b). HH: [2][1024][256] bf16.
__global__ void enc_lstm_k(const float* __restrict__ gx, const bf16* __restrict__ HH,
                           const float* __restrict__ bias, bf16* out){
  extern __shared__ char sm[];
  float* zsh = (float*)sm;                         // [16][1024]
  float* csh = (float*)(sm + 65536);               // [16][256]
  bf16*  hsh = (bf16*)(sm + 65536 + 16384);        // [16][256]
  int dir = blockIdx.x;
  const float* gxd = gx + (size_t)dir*2048*1024;
  const bf16*  HHd = HH + (size_t)dir*1024*256;
  const float* bd  = bias + dir*1024;
  int tid = threadIdx.x;
  for (int i = tid; i < 16*256; i += 1024){ csh[i] = 0.f; hsh[i] = (bf16)0.f; }
  __syncthreads();
  int wv = tid >> 5, lane = tid & 31;
  int n = lane & 15, mb = (lane & 16) ? 8 : 0;
  for (int step = 0; step < 128; ++step){
    int t = dir ? (127 - step) : step;
#pragma unroll
    for (int u2 = 0; u2 < 2; ++u2){
      int nt = wv*2 + u2;
      v8f acc = ZV8F;
      const bf16* Wb = HHd + (size_t)nt*16*256;
      for (int k = 0; k < 256; k += 32)
        acc = WMMA_BF(ld_tile(hsh + k, 256), ld_tile(Wb + k, 256), acc);
#pragma unroll
      for (int r = 0; r < 8; ++r) zsh[(mb + r)*1024 + nt*16 + n] = acc[r];
    }
    __syncthreads();
    for (int e = tid; e < 4096; e += 1024){
      int bb = e >> 8, u = e & 255;
      const float* gr = gxd + ((size_t)(t*16 + bb))*1024;
      float zi = zsh[bb*1024 + u]       + gr[u]       + bd[u];
      float zf = zsh[bb*1024 + 256 + u] + gr[256 + u] + bd[256 + u];
      float zg = zsh[bb*1024 + 512 + u] + gr[512 + u] + bd[512 + u];
      float zo = zsh[bb*1024 + 768 + u] + gr[768 + u] + bd[768 + u];
      float c0 = csh[bb*256 + u];
      float nc = sigm(zf)*c0 + sigm(zi)*tanhf(zg);
      float h  = sigm(zo)*tanhf(nc);
      csh[bb*256 + u] = nc;
      hsh[bb*256 + u] = (bf16)h;
      out[((size_t)(t*16 + bb))*512 + (dir ? 256 : 0) + u] = (bf16)h;
    }
    __syncthreads();
  }
}

// ---------------- attention (one WG per batch element) ----------------
__global__ void attn_k(const float* h2F, const float* awP, float* awC,
                       const float* mem, const bf16* encB,
                       const float* attq, const float* locw, const float* locb,
                       const float* attl, const float* atte, const float* attb,
                       float* attnOut, bf16* ctxOut, int tstep){
  int b = blockIdx.x, tid = threadIdx.x;
  __shared__ float q[128];
  __shared__ float fi[32][128];
  __shared__ float ebuf[128];
  __shared__ float sinv;
  { const float* hb = h2F + b*1024;
    const float* qr = attq + (size_t)tid*1024;
    float s = 0.f;
    for (int d2 = 0; d2 < 1024; ++d2) s += hb[d2]*qr[d2];
    q[tid] = s; }
  for (int e = tid; e < 32*128; e += 128){
    int cc = e >> 7, t = e & 127;
    float a = locb[cc];
    for (int j = 0; j < 31; ++j){
      int tt = t + j - 15;
      if (tt >= 0 && tt < 128) a += awP[b*128 + tt]*locw[cc*31 + j];
    }
    fi[cc][t] = a;
  }
  __syncthreads();
  { int t = tid;
    const float* mr = mem + ((size_t)(t*16 + b))*128;
    float et = 0.f;
    for (int k = 0; k < 128; ++k){
      float l = 0.f;
      for (int cc = 0; cc < 32; ++cc) l += fi[cc][t]*attl[k*32 + cc];
      et += tanhf(q[k] + mr[k] + l + attb[k]) * atte[k];
    }
    ebuf[t] = et; }
  __syncthreads();
  if (tid == 0){
    float m = -1e30f;
    for (int i = 0; i < 128; ++i) m = fmaxf(m, ebuf[i]);
    float ss = 0.f;
    for (int i = 0; i < 128; ++i){ float e2 = __expf(ebuf[i] - m); ebuf[i] = e2; ss += e2; }
    sinv = 1.f/ss;
  }
  __syncthreads();
  float w = ebuf[tid]*sinv;
  ebuf[tid] = w;
  awC[b*128 + tid] = w;
  attnOut[(size_t)b*51200 + (size_t)tstep*128 + tid] = w;
  __syncthreads();
  for (int c = tid; c < 512; c += 128){
    float cs = 0.f;
    for (int t = 0; t < 128; ++t) cs += ebuf[t]*(float)encB[((size_t)(t*16 + b))*512 + c];
    ctxOut[b*512 + c] = (bf16)cs;
  }
}

// ---------------- decoder LSTM cell step: 64 WGs, each owns 16 hidden units ----------------
// z = [A0|A1|A2] @ W^T; wave g computes gate-block g's 16x16 tile; cell update after barrier.
__global__ void dec_lstm_k(const bf16* A0, int ld0, int K0,
                           const bf16* A1, int ld1, int K1,
                           const bf16* A2, int ld2,
                           const bf16* __restrict__ W, int Ktot, const float* bias,
                           float* c, float* hF, bf16* hB, bf16* hB2){
  __shared__ float zsh[4][16][16];
  int j = blockIdx.x;
  int wv = threadIdx.x >> 5, lane = threadIdx.x & 31;
  const bf16* Wb = W + (size_t)(wv*1024 + j*16)*Ktot;
  v8f acc = ZV8F;
  for (int k = 0; k < Ktot; k += 32){
    int kk = k; const bf16* Ab; int ld;
    if (kk < K0){ Ab = A0; ld = ld0; }
    else { kk -= K0; if (kk < K1){ Ab = A1; ld = ld1; } else { kk -= K1; Ab = A2; ld = ld2; } }
    acc = WMMA_BF(ld_tile(Ab + kk, ld), ld_tile(Wb + k, Ktot), acc);
  }
  int n = lane & 15, mb = (lane & 16) ? 8 : 0;
#pragma unroll
  for (int r = 0; r < 8; ++r) zsh[wv][mb + r][n] = acc[r];
  __syncthreads();
  for (int e = threadIdx.x; e < 256; e += 128){
    int bb = e >> 4, u = e & 15;
    int col = j*16 + u;
    float zi = zsh[0][bb][u] + bias[col];
    float zf = zsh[1][bb][u] + bias[1024 + col];
    float zg = zsh[2][bb][u] + bias[2048 + col];
    float zo = zsh[3][bb][u] + bias[3072 + col];
    float c0 = c[bb*1024 + col];
    float nc = sigm(zf)*c0 + sigm(zi)*tanhf(zg);
    float h  = sigm(zo)*tanhf(nc);
    c[bb*1024 + col] = nc;
    if (hF) hF[bb*1024 + col] = h;
    hB[bb*1024 + col] = (bf16)h;
    if (hB2) hB2[bb*1024 + col] = (bf16)h;
  }
}

// ---------------- mel/stop projection: [h2|ctx][1536] @ msW[96][1536] ----------------
__global__ void melstop_k(const bf16* h2all, const bf16* ctx, const bf16* __restrict__ W,
                          const float* bias, float* melOut, float* stopOut, bf16* melB){
  int nt = blockIdx.x, mt = blockIdx.y;
  const bf16* Wb = W + (size_t)nt*16*1536;
  v8f acc = ZV8F;
  for (int k = 0; k < 1536; k += 32){
    const bf16* Ab; int ld, kk;
    if (k < 1024){ Ab = h2all + (size_t)mt*16*1024; ld = 1024; kk = k; }
    else         { Ab = ctx   + (size_t)mt*16*512;  ld = 512;  kk = k - 1024; }
    acc = WMMA_BF(ld_tile(Ab + kk, ld), ld_tile(Wb + k, 1536), acc);
  }
  int lane = threadIdx.x;
  int n = lane & 15, mb = (lane & 16) ? 8 : 0;
  int o = nt*16 + n;
  float bs = bias[o];
#pragma unroll
  for (int r = 0; r < 8; ++r){
    int m = mt*16 + mb + r;
    int t = m >> 4, b = m & 15;
    float v = acc[r] + bs;
    if (o < 80){
      melOut[((size_t)b*400 + t)*80 + o] = v;
      melB[((size_t)b*404 + 2 + t)*96 + o] = (bf16)v;
    } else if (o == 80) stopOut[(size_t)b*400 + t] = v;
  }
}

// ---------------- WaveNet gated residual layer ----------------
// H,Hn: [16][912][64] bf16 (512 zero rows of causal padding per batch).
__global__ void wn_layer_k(const bf16* __restrict__ H, bf16* Hn,
                           const bf16* FW, const bf16* GW,
                           const bf16* RW, const bf16* SW,
                           const float* fb, const float* gb,
                           const float* rb, const float* sb,
                           float* skip, int d){
  __shared__ __attribute__((aligned(16))) bf16 obuf[16][64];
  int mt = blockIdx.x;
  int b = mt / 25, t0 = (mt % 25) << 4;
  int wv = threadIdx.x >> 5, lane = threadIdx.x & 31;
  int n = lane & 15, mb = (lane & 16) ? 8 : 0;
  int col = wv*16 + n;
  size_t rowT = (size_t)b*912 + 512 + t0;
  size_t rowD = rowT - d;
  const bf16* fw0 = FW + (size_t)wv*16*64;
  const bf16* fw1 = FW + (size_t)(64 + wv*16)*64;
  const bf16* gw0 = GW + (size_t)wv*16*64;
  const bf16* gw1 = GW + (size_t)(64 + wv*16)*64;
  v8f fa = ZV8F, ga = ZV8F;
  for (int k = 0; k < 64; k += 32){
    v16bf a0 = ld_tile(H + rowD*64 + k, 64);
    v16bf a1 = ld_tile(H + rowT*64 + k, 64);
    fa = WMMA_BF(a0, ld_tile(fw0 + k, 64), fa);
    fa = WMMA_BF(a1, ld_tile(fw1 + k, 64), fa);
    ga = WMMA_BF(a0, ld_tile(gw0 + k, 64), ga);
    ga = WMMA_BF(a1, ld_tile(gw1 + k, 64), ga);
  }
  float fbc = fb[col], gbc = gb[col];
#pragma unroll
  for (int r = 0; r < 8; ++r)
    obuf[mb + r][col] = (bf16)(tanhf(fa[r] + fbc) * sigm(ga[r] + gbc));
  __syncthreads();
  const bf16* swr = SW + (size_t)wv*16*64;
  const bf16* rwr = RW + (size_t)wv*16*64;
  v8f sa = ZV8F, ra = ZV8F;
  for (int k = 0; k < 64; k += 32){
    sa = WMMA_BF(ld_tile(&obuf[0][0] + k, 64), ld_tile(swr + k, 64), sa);
    ra = WMMA_BF(ld_tile(H + rowT*64 + k, 64), ld_tile(rwr + k, 64), ra);
  }
  float sbc = sb[col], rbc = rb[col];
#pragma unroll
  for (int r = 0; r < 8; ++r){
    size_t m = (size_t)mt*16 + mb + r;
    skip[m*64 + col] += sa[r] + sbc;
    float hv = (float)H[(rowT + mb + r)*64 + col];
    Hn[(rowT + mb + r)*64 + col] = (bf16)(hv + ra[r] + rbc);
  }
}

// ---------------- WaveNet output head ----------------
__global__ void wnout_k(const float* skip, const float* o1w, const float* o1b,
                        const float* o2w, const float* o2b, float* voc){
  __shared__ float s1[64], s2[64];
  int m = blockIdx.x, tid = threadIdx.x;
  int b = m / 400, t = m % 400;
  s1[tid] = fmaxf(skip[(size_t)m*64 + tid], 0.f);
  __syncthreads();
  float a = o1b[tid];
  for (int k = 0; k < 64; ++k) a += o1w[tid*64 + k]*s1[k];
  s2[tid] = fmaxf(a, 0.f);
  __syncthreads();
  for (int cc = tid; cc < 256; cc += 64){
    float v = o2b[cc];
    for (int k = 0; k < 64; ++k) v += o2w[cc*64 + k]*s2[k];
    voc[((size_t)b*256 + cc)*400 + t] = v;
  }
}

// ======================= host =======================
static inline int eb(long n){ long b = (n + 255)/256; return (int)(b > 4096 ? 4096 : (b < 1 ? 1 : b)); }
static inline void zfill(void* p, size_t bytes, hipStream_t s){
  long n = (long)(bytes/4);
  zero_k<<<eb(n),256,0,s>>>((unsigned*)p, n);
}

extern "C" void kernel_launch(void* const* d_in, const int* in_sizes, int n_in,
                              void* d_out, int out_size, void* d_ws, size_t ws_size,
                              hipStream_t stream){
  (void)in_sizes; (void)n_in; (void)out_size; (void)ws_size;
#define FI(i) ((const float*)d_in[i])
  const int* Xids = (const int*)d_in[0];
  const float* MEL = FI(1);

  // output regions
  float* o_mel  = (float*)d_out;       // [16][400][80]
  float* o_post = o_mel + 512000;      // [16][400][80]
  float* o_stop = o_post + 512000;     // [16][400]
  float* o_attn = o_stop + 6400;       // [16][400][128]
  float* o_voc  = o_attn + 819200;     // [16][256][400]

  // workspace bump allocator
  char* base = (char*)d_ws; size_t off = 0;
  auto alloc = [&](size_t bytes)->char*{ char* r = base + off; off = (off + bytes + 255) & ~(size_t)255; return r; };

  // packed bf16 weights
  bf16* encW  = (bf16*)alloc((size_t)3*5*512*512*2);
  bf16* encIH = (bf16*)alloc((size_t)4*1024*512*2);
  bf16* encHH = (bf16*)alloc((size_t)4*1024*256*2);
  float* encLB = (float*)alloc(4096*4);
  bf16* decW1 = (bf16*)alloc((size_t)4096*1792*2);
  bf16* decW2 = (bf16*)alloc((size_t)4096*2048*2);
  float* decB1 = (float*)alloc(4096*4);
  float* decB2 = (float*)alloc(4096*4);
  bf16* preW1 = (bf16*)alloc((size_t)256*96*2);
  bf16* preW2 = (bf16*)alloc((size_t)256*256*2);
  bf16* attM  = (bf16*)alloc((size_t)128*512*2);
  bf16* msW   = (bf16*)alloc((size_t)96*1536*2);
  float* msB  = (float*)alloc(96*4);
  bf16* pnW   = (bf16*)alloc((size_t)4423680*2);
  float* pnCB = (float*)alloc(5*512*4);
  float* pnSC = (float*)alloc(5*512*4);
  float* pnBT = (float*)alloc(5*512*4);
  float* encCB = (float*)alloc(3*512*4);
  float* encSC = (float*)alloc(3*512*4);
  float* encBT = (float*)alloc(3*512*4);
  bf16* wnFW = (bf16*)alloc((size_t)20*2*64*64*2);
  bf16* wnGW = (bf16*)alloc((size_t)20*2*64*64*2);
  bf16* wnRW = (bf16*)alloc((size_t)20*64*64*2);
  bf16* wnSW = (bf16*)alloc((size_t)20*64*64*2);
  bf16* wnInW = (bf16*)alloc((size_t)64*96*2);

  // activations
  bf16* embx  = (bf16*)alloc((size_t)16*132*512*2);
  bf16* cvA   = (bf16*)alloc((size_t)16*132*512*2);
  bf16* cvB   = (bf16*)alloc((size_t)16*132*512*2);
  bf16* xs0   = (bf16*)alloc((size_t)2048*512*2);
  bf16* xsMid = (bf16*)alloc((size_t)2048*512*2);
  bf16* encBf = (bf16*)alloc((size_t)2048*512*2);
  float* gx   = (float*)alloc((size_t)2*2048*1024*4);
  float* mem  = (float*)alloc((size_t)2048*128*4);
  bf16* tf    = (bf16*)alloc((size_t)6400*96*2);
  bf16* pr1   = (bf16*)alloc((size_t)6400*256*2);
  bf16* prB   = (bf16*)alloc((size_t)6400*256*2);
  bf16* ctxB  = (bf16*)alloc((size_t)6400*512*2);
  bf16* h2all = (bf16*)alloc((size_t)6400*1024*2);
  float* c1   = (float*)alloc(16*1024*4);
  float* c2   = (float*)alloc(16*1024*4);
  bf16* h1a   = (bf16*)alloc(16*1024*2);
  bf16* h1b   = (bf16*)alloc(16*1024*2);
  bf16* h2a   = (bf16*)alloc(16*1024*2);
  bf16* h2b   = (bf16*)alloc(16*1024*2);
  float* h2F  = (float*)alloc(16*1024*4);
  float* aw0  = (float*)alloc(16*128*4);
  float* aw1  = (float*)alloc(16*128*4);
  bf16* melB  = (bf16*)alloc((size_t)16*404*96*2);
  bf16* mpB   = (bf16*)alloc((size_t)16*404*96*2);
  bf16* pnA   = (bf16*)alloc((size_t)16*404*512*2);
  bf16* pnB   = (bf16*)alloc((size_t)16*404*512*2);
  bf16* wnA   = (bf16*)alloc((size_t)16*912*64*2);
  bf16* wnB   = (bf16*)alloc((size_t)16*912*64*2);
  float* skip = (float*)alloc((size_t)6400*64*4);

  // ---- zero state / padded buffers (every call, deterministic) ----
  zfill(embx, (size_t)16*132*512*2, stream);
  zfill(cvA,  (size_t)16*132*512*2, stream);
  zfill(cvB,  (size_t)16*132*512*2, stream);
  zfill(melB, (size_t)16*404*96*2, stream);
  zfill(mpB,  (size_t)16*404*96*2, stream);
  zfill(pnA,  (size_t)16*404*512*2, stream);
  zfill(pnB,  (size_t)16*404*512*2, stream);
  zfill(wnA,  (size_t)16*912*64*2, stream);
  zfill(wnB,  (size_t)16*912*64*2, stream);
  zfill(skip, (size_t)6400*64*4, stream);
  zfill(c1, 16*1024*4, stream); zfill(c2, 16*1024*4, stream);
  zfill(h1a, 16*1024*2, stream); zfill(h2a, 16*1024*2, stream);
  zfill(h2F, 16*1024*4, stream); zfill(aw0, 16*128*4, stream);

  // ---- weight packing ----
  for (int l = 0; l < 3; ++l){
    packconv_k<<<eb(1310720),256,0,stream>>>(encW + (size_t)l*5*512*512, FI(3) + (size_t)l*512*512*5, 512,512,5,512,512);
    convsc_k<<<2,256,0,stream>>>(encCB + l*512, encSC + l*512, encBT + l*512, FI(4)+l*512, FI(5)+l*512, FI(6)+l*512, 512, 512);
  }
  cvt_k<<<eb(2097152),256,0,stream>>>(encIH, FI(7), 2097152);
  cvt_k<<<eb(1048576),256,0,stream>>>(encHH, FI(8), 1048576);
  addb_k<<<eb(4096),256,0,stream>>>(encLB, FI(9), FI(10), 4096);
  cvtpad_k<<<eb(24576),256,0,stream>>>(preW1, FI(11), 256, 80, 96);
  cvt_k<<<eb(65536),256,0,stream>>>(preW2, FI(13), 65536);
  concat2_k<<<4096,256,0,stream>>>(decW1, FI(15), 768, FI(16), 1024, 4096);
  concat2_k<<<4096,256,0,stream>>>(decW2, FI(19), 1024, FI(20), 1024, 4096);
  addb_k<<<eb(4096),256,0,stream>>>(decB1, FI(17), FI(18), 4096);
  addb_k<<<eb(4096),256,0,stream>>>(decB2, FI(21), FI(22), 4096);
  cvt_k<<<eb(65536),256,0,stream>>>(attM, FI(24), 65536);
  packms_k<<<eb(147456),256,0,stream>>>(msW, msB, FI(30), FI(31), FI(32), FI(33));
  {
    const size_t pnWoff[5] = {0, 245760, 245760+1310720, 245760+2*1310720, 245760+3*1310720};
    const int pnOr[5] = {512,512,512,512,80};
    const int pnIr[5] = {80,512,512,512,512};
    const int pnOp[5] = {512,512,512,512,96};
    const int pnIp[5] = {96,512,512,512,512};
    for (int l = 0; l < 5; ++l){
      long n = (long)5*pnOp[l]*pnIp[l];
      packconv_k<<<eb(n),256,0,stream>>>(pnW + pnWoff[l], FI(34+l), pnOr[l], pnIr[l], 5, pnOp[l], pnIp[l]);
      convsc_k<<<2,256,0,stream>>>(pnCB + l*512, pnSC + l*512, pnBT + l*512, FI(39+l), FI(44+l), FI(49+l), pnOr[l], pnOp[l]);
    }
  }
  for (int l = 0; l < 20; ++l){
    packconv_k<<<eb(8192),256,0,stream>>>(wnFW + (size_t)l*8192, FI(56) + (size_t)l*8192, 64,64,2,64,64);
    packconv_k<<<eb(8192),256,0,stream>>>(wnGW + (size_t)l*8192, FI(58) + (size_t)l*8192, 64,64,2,64,64);
  }
  cvt_k<<<eb(81920),256,0,stream>>>(wnRW, FI(60), 81920);
  cvt_k<<<eb(81920),256,0,stream>>>(wnSW, FI(62), 81920);
  cvtpad_k<<<eb(6144),256,0,stream>>>(wnInW, FI(54), 64, 80, 96);

  // ---- encoder ----
  embed_k<<<eb(1048576),256,0,stream>>>(embx, Xids, FI(2));
  dim3 ecg(32,128);
  conv5_k<<<ecg,32,0,stream>>>(embx, encW,                     encCB,      encSC,      encBT,      512,512,128,1, cvA, nullptr,nullptr,0);
  conv5_k<<<ecg,32,0,stream>>>(cvA, encW + (size_t)5*512*512,  encCB+512,  encSC+512,  encBT+512,  512,512,128,1, cvB, nullptr,nullptr,0);
  conv5_k<<<ecg,32,0,stream>>>(cvB, encW + (size_t)10*512*512, encCB+1024, encSC+1024, encBT+1024, 512,512,128,0, cvA, nullptr,nullptr,0);
  stage_xs_k<<<eb(1048576),256,0,stream>>>(xs0, cvA);

  for (int l = 0; l < 2; ++l){
    const bf16* xin = l ? xsMid : xs0;
    bf16* xout = l ? encBf : xsMid;
    for (int d = 0; d < 2; ++d)
      gemm_k<<<dim3(64,128),32,0,stream>>>(xin,512, encIH + (size_t)(l*2+d)*1024*512, nullptr, 512,0,
                                           gx + (size_t)d*2048*1024,1024, nullptr,0, 2048,0,0, 2048,0,0);
    enc_lstm_k<<<2,1024,90112,stream>>>(gx, encHH + (size_t)l*2*1024*256, encLB + l*2*1024, xout);
  }
  gemm_k<<<dim3(8,128),32,0,stream>>>(encBf,512, attM, nullptr, 512,0, mem,128, nullptr,0, 2048,0,0, 2048,0,0);

  // ---- prenet (batched over all 400 steps: teacher-forced input) ----
  stage_tf_k<<<eb(614400),256,0,stream>>>(tf, MEL);
  gemm_k<<<dim3(16,400),32,0,stream>>>(tf,96,  preW1, FI(12), 96,1,  nullptr,0, pr1,256, 6400,0,0, 6400,0,0);
  gemm_k<<<dim3(16,400),32,0,stream>>>(pr1,256, preW2, FI(14), 256,1, nullptr,0, prB,256, 6400,0,0, 6400,0,0);

  // ---- decoder: 400 sequential steps ----
  for (int t = 0; t < 400; ++t){
    const float* awP = (t & 1) ? aw1 : aw0;
    float* awC = (t & 1) ? aw0 : aw1;
    const bf16* h1P = (t & 1) ? h1b : h1a;  bf16* h1C = (t & 1) ? h1a : h1b;
    const bf16* h2P = (t & 1) ? h2b : h2a;  bf16* h2C = (t & 1) ? h2a : h2b;
    attn_k<<<16,128,0,stream>>>(h2F, awP, awC, mem, encBf,
                                FI(23), FI(26), FI(27), FI(25), FI(28), FI(29),
                                o_attn, ctxB + (size_t)t*8192, t);
    dec_lstm_k<<<64,128,0,stream>>>(prB + (size_t)t*4096, 256, 256,
                                    ctxB + (size_t)t*8192, 512, 512,
                                    h1P, 1024,
                                    decW1, 1792, decB1, c1, nullptr, h1C, nullptr);
    dec_lstm_k<<<64,128,0,stream>>>(h1C, 1024, 1024,
                                    h2P, 1024, 1024,
                                    nullptr, 0,
                                    decW2, 2048, decB2, c2, h2F, h2C, h2all + (size_t)t*16384);
  }
  melstop_k<<<dim3(6,400),32,0,stream>>>(h2all, ctxB, msW, msB, o_mel, o_stop, melB);

  // ---- postnet ----
  {
    const size_t pnWoff[5] = {0, 245760, 245760+1310720, 245760+2*1310720, 245760+3*1310720};
    conv5_k<<<dim3(32,400),32,0,stream>>>(melB, pnW+pnWoff[0], pnCB,        pnSC,        pnBT,        96, 512,400,2, pnA, nullptr,nullptr,0);
    conv5_k<<<dim3(32,400),32,0,stream>>>(pnA,  pnW+pnWoff[1], pnCB+512,    pnSC+512,    pnBT+512,    512,512,400,2, pnB, nullptr,nullptr,0);
    conv5_k<<<dim3(32,400),32,0,stream>>>(pnB,  pnW+pnWoff[2], pnCB+1024,   pnSC+1024,   pnBT+1024,   512,512,400,2, pnA, nullptr,nullptr,0);
    conv5_k<<<dim3(32,400),32,0,stream>>>(pnA,  pnW+pnWoff[3], pnCB+1536,   pnSC+1536,   pnBT+1536,   512,512,400,2, pnB, nullptr,nullptr,0);
    conv5_k<<<dim3(6,400),32,0,stream>>>( pnB,  pnW+pnWoff[4], pnCB+2048,   pnSC+2048,   pnBT+2048,   512, 96,400,0, mpB, o_mel, o_post, 80);
  }

  // ---- wavenet vocoder ----
  gemm_k<<<dim3(4,400),32,0,stream>>>(mpB,96, wnInW, FI(55), 96,0, nullptr,0, wnA,64, 400,4,2, 400,512,512);
  {
    const bf16* hc = wnA; bf16* hn = wnB;
    for (int l = 0; l < 20; ++l){
      int d = 1 << (l % 10);
      wn_layer_k<<<400,128,0,stream>>>(hc, hn, wnFW + (size_t)l*8192, wnGW + (size_t)l*8192,
                                       wnRW + (size_t)l*4096, wnSW + (size_t)l*4096,
                                       FI(57)+l*64, FI(59)+l*64, FI(61)+l*64, FI(63)+l*64, skip, d);
      const bf16* tmp = hn; hn = (bf16*)hc; hc = tmp;
    }
  }
  wnout_k<<<6400,64,0,stream>>>(skip, FI(64), FI(65), FI(66), FI(67), o_voc);
#undef FI
}
